// AgentAttention_12343736008877
// MI455X (gfx1250) — matrix operand order repaired and checked by reading, not compiled, not run on hardware
//
#include <hip/hip_runtime.h>
#include <hip/hip_bf16.h>
#include <stdint.h>

// ---------------------------------------------------------------------------
// Agent Attention (b=16, n=56*56, dim=512, heads=8, dh=64, agents=49) for
// gfx1250. All matmuls run on v_wmma_f32_16x16x32_bf16; softmax/bias in f32.
// ---------------------------------------------------------------------------

typedef __attribute__((ext_vector_type(16))) __bf16 v16bf;
typedef __attribute__((ext_vector_type(8)))  float  v8f;
typedef unsigned short u16;
typedef unsigned int   u32;

#define BATCH 16
#define HH    56
#define WW    56
#define NTOK  3136          // 56*56
#define DIM   512
#define IDIM  512           // heads*dh
#define HEADS 8
#define DH    64
#define AG    49
#define AGP   64            // padded agent count
#define NQKV  1536          // q|k|v columns
#define MROWS (BATCH*NTOK)  // 50176
#define SCALE 0.125f        // dh^-0.5

__device__ __forceinline__ u16 f2bf(float f) {
  u32 u = __float_as_uint(f);
  u32 r = (u + 0x7FFFu + ((u >> 16) & 1u)) >> 16;   // round-to-nearest-even
  return (u16)r;
}
__device__ __forceinline__ float bf2f(u16 h) {
  return __uint_as_float(((u32)h) << 16);
}
__device__ __forceinline__ int imin(int a, int b) { return a < b ? a : b; }

union FragBF { v16bf v; uint4 q[2]; u16 h[16]; };

// A-matrix (16x32 bf16, MxK) fragment. base -> (m=0,k=0) of the tile region,
// rowStride in halfs. ISA layout: lane m = lane%16; elements 0..7 hold
// K = kOff + 8*(lane/16) + e ; elements 8..15 hold K = that + 16.
__device__ __forceinline__ v16bf loadAfrag(const u16* base, int rowStride,
                                           int lane, int kOff) {
  int m  = lane & 15;
  int kb = kOff + ((lane >> 4) << 3);
  FragBF f;
  f.q[0] = *(const uint4*)(base + (size_t)m * rowStride + kb);
  f.q[1] = *(const uint4*)(base + (size_t)m * rowStride + kb + 16);
  return f.v;
}

// B-matrix (32x16 bf16, KxN) fragment from N-major (transposed) storage:
// base -> row n=0 (each row holds K contiguous). lane col n = lane%16;
// elements 0..7 hold K = kOff + 8*(lane/16) + e ; elements 8..15 hold K = that + 16.
__device__ __forceinline__ v16bf loadBfragT(const u16* base, int rowStride,
                                            int lane, int kOff) {
  int n  = lane & 15;
  int k0 = kOff + ((lane >> 4) << 3);
  FragBF f;
  f.q[0] = *(const uint4*)(base + (size_t)n * rowStride + k0);
  f.q[1] = *(const uint4*)(base + (size_t)n * rowStride + k0 + 16);
  return f.v;
}

__device__ __forceinline__ v8f wmma_bf16(v16bf a, v16bf b, v8f c) {
  return __builtin_amdgcn_wmma_f32_16x16x32_bf16(false, a, false, b,
                                                 (short)0, c, false, false);
}

// ------------------------------ pack kernels -------------------------------

__global__ void pack_x(const float* __restrict__ x, u16* __restrict__ xb,
                       size_t nElem) {
  size_t i = (size_t)blockIdx.x * blockDim.x + threadIdx.x;
  if (i < nElem) xb[i] = f2bf(x[i]);
}

__global__ void pack_w(const float* __restrict__ qw, const float* __restrict__ qb,
                       const float* __restrict__ kvw, const float* __restrict__ kvb,
                       const float* __restrict__ pw,
                       u16* __restrict__ wqkv, u16* __restrict__ wproj,
                       float* __restrict__ bqkv) {
  int i = blockIdx.x * blockDim.x + threadIdx.x;
  if (i < DIM * NQKV) {
    int k = i / NQKV, j = i % NQKV;
    float v = (j < IDIM) ? qw[k * IDIM + j] : kvw[k * (2 * IDIM) + (j - IDIM)];
    wqkv[i] = f2bf(v);
  }
  if (i < DIM * DIM) wproj[i] = f2bf(pw[i]);
  if (i < NQKV) bqkv[i] = (i < IDIM) ? qb[i] : kvb[i - IDIM];
}

// ------------------------- generic 64x64-tile GEMM -------------------------
// C(MxN) = A(MxK,bf16) @ W(KxN,bf16) + bias ; out bf16 or f32 (compile-time).
// 128 threads = 4 waves; wave w computes rows [16w,16w+16) x 64 cols.
// Double-buffered LDS stage: one barrier per k-step; next global tile is
// register-prefetched while the four WMMAs of the current step execute.

template <int N, int K, bool OUTF32>
__global__ __launch_bounds__(128) void gemm64(
    const u16* __restrict__ A, const u16* __restrict__ W,
    const float* __restrict__ bias, void* __restrict__ out)
{
  __shared__ __align__(16) u16 As[2][64 * 40];
  __shared__ __align__(16) u16 Bs[2][64 * 40];
  const int tid   = threadIdx.x;
  const int lane  = tid & 31;
  const int wv    = tid >> 5;
  const int mBase = blockIdx.x * 64;
  const int nBase = blockIdx.y * 64;

  const int aRow = tid >> 1;            // 0..63
  const int aCol = (tid & 1) << 4;      // 0,16
  const int bK   = tid >> 2;            // 0..31
  const int bN   = (tid & 3) << 4;      // 0,16,32,48

  const u16* gA = A + (size_t)(mBase + aRow) * K + aCol;
  const u16* gB = W + (size_t)bK * N + nBase + bN;

  uint4 ra0 = *(const uint4*)gA;
  uint4 ra1 = *(const uint4*)(gA + 8);
  uint4 rb0 = *(const uint4*)gB;
  uint4 rb1 = *(const uint4*)(gB + 8);

  v8f acc0 = {}, acc1 = {}, acc2 = {}, acc3 = {};
  constexpr int NK = K / 32;
  int buf = 0;
  #pragma unroll 2
  for (int ks = 0; ks < NK; ++ks) {
    *(uint4*)(&As[buf][aRow * 40 + aCol])     = ra0;
    *(uint4*)(&As[buf][aRow * 40 + aCol + 8]) = ra1;
    {
      FragBF f; f.q[0] = rb0; f.q[1] = rb1;
      #pragma unroll
      for (int i = 0; i < 16; ++i) Bs[buf][(bN + i) * 40 + bK] = f.h[i];
    }
    __syncthreads();
    if (ks + 1 < NK) {                  // prefetch next tile into registers
      gA += 32;
      gB += (size_t)32 * N;
      ra0 = *(const uint4*)gA;
      ra1 = *(const uint4*)(gA + 8);
      rb0 = *(const uint4*)gB;
      rb1 = *(const uint4*)(gB + 8);
      __builtin_prefetch(gA + 32, 0, 1);            // global_prefetch_b8
      __builtin_prefetch(gB + (size_t)32 * N, 0, 1);
    }
    v16bf av = loadAfrag(&As[buf][(wv * 16) * 40], 40, lane, 0);
    v16bf b0 = loadBfragT(&Bs[buf][0],       40, lane, 0);
    v16bf b1 = loadBfragT(&Bs[buf][16 * 40], 40, lane, 0);
    v16bf b2 = loadBfragT(&Bs[buf][32 * 40], 40, lane, 0);
    v16bf b3 = loadBfragT(&Bs[buf][48 * 40], 40, lane, 0);
    acc0 = wmma_bf16(av, b0, acc0);
    acc1 = wmma_bf16(av, b1, acc1);
    acc2 = wmma_bf16(av, b2, acc2);
    acc3 = wmma_bf16(av, b3, acc3);
    buf ^= 1;
  }

  v8f accs[4] = {acc0, acc1, acc2, acc3};
  int nloc = lane & 15;
  int mloc = wv * 16 + ((lane >> 4) << 3);
  #pragma unroll
  for (int nt = 0; nt < 4; ++nt) {
    int n = nBase + nt * 16 + nloc;
    float bv = bias[n];
    #pragma unroll
    for (int r = 0; r < 8; ++r) {
      size_t idx = (size_t)(mBase + mloc + r) * N + n;
      float v = accs[nt][r] + bv;
      if (OUTF32) ((float*)out)[idx] = v;
      else        ((u16*)out)[idx]   = f2bf(v);
    }
  }
}

// ------------------------- agent pooling (mean 8x8) ------------------------
// agent_bf[b][h][a][d] = scale * mean over 8x8 block of q ; zero pad a>=49.

__global__ void agent_pool(const u16* __restrict__ qkv, u16* __restrict__ agent)
{
  int a = blockIdx.x & 63;
  int b = blockIdx.x >> 6;
  for (int c = threadIdx.x; c < IDIM; c += blockDim.x) {
    int h = c >> 6, d = c & 63;
    float s = 0.f;
    if (a < AG) {
      int ay = a / 7, ax = a % 7;
      for (int yy = 0; yy < 8; ++yy)
        for (int xx = 0; xx < 8; ++xx) {
          int n = (ay * 8 + yy) * WW + (ax * 8 + xx);
          s += bf2f(qkv[(size_t)(b * NTOK + n) * NQKV + c]);
        }
      s *= (1.f / 64.f) * SCALE;
    }
    agent[(size_t)((b * HEADS + h) * AGP + a) * DH + d] = f2bf(s);
  }
}

// ------------------------------ bias tables --------------------------------

__device__ __forceinline__ void lin_w(int o, int& i0, int& i1, float& w) {
  float src = (o + 0.5f) * 0.125f - 0.5f;       // 7/56 scale
  src = fmaxf(src, 0.f);
  i0 = imin((int)floorf(src), 6);
  i1 = imin(i0 + 1, 6);
  w  = src - (float)i0;
}

__device__ __forceinline__ float bilin7(const float* p, int y, int x) {
  int y0, y1, x0, x1; float wy, wx;
  lin_w(y, y0, y1, wy);
  lin_w(x, x0, x1, wx);
  float t0 = p[y0 * 7 + x0] * (1.f - wy) + p[y1 * 7 + x0] * wy;
  float t1 = p[y0 * 7 + x1] * (1.f - wy) + p[y1 * 7 + x1] * wy;
  return t0 * (1.f - wx) + t1 * wx;
}

// pb[h][a][n] = bilinear(an_bias)+ah+aw ; 0 for pad rows.
__global__ void bias_pb(const float* __restrict__ an, const float* __restrict__ ahb,
                        const float* __restrict__ awb, float* __restrict__ pb)
{
  int a = blockIdx.x & 63;
  int h = blockIdx.x >> 6;
  for (int n = threadIdx.x; n < NTOK; n += blockDim.x) {
    float v = 0.f;
    if (a < AG) {
      int y = n / WW, x = n % WW;
      v = bilin7(an + (size_t)(h * AG + a) * 49, y, x)
        + ahb[(h * AG + a) * HH + y] + awb[(h * AG + a) * WW + x];
    }
    pb[(size_t)(h * AGP + a) * NTOK + n] = v;
  }
}

// nb[h][n][a] = bilinear(na_bias)+ha+wa ; -1e30 for pad cols (softmax -> 0).
__global__ void bias_nb(const float* __restrict__ na, const float* __restrict__ hab,
                        const float* __restrict__ wab, float* __restrict__ nb)
{
  int n = blockIdx.x % NTOK;
  int h = blockIdx.x / NTOK;
  int a = threadIdx.x;
  float v = -1e30f;
  if (a < AG) {
    int y = n / WW, x = n % WW;
    v = bilin7(na + (size_t)(h * AG + a) * 49, y, x)
      + hab[(h * HH + y) * AG + a] + wab[(h * WW + x) * AG + a];
  }
  nb[(size_t)(h * NTOK + n) * AGP + a] = v;
}

// ---------------------- s1 = (scale*agent) @ k^T + pb ----------------------
// grid (196 ntiles, b*h); 4 waves = 4 a-tiles of 16.

__global__ __launch_bounds__(128) void attn_s1(
    const u16* __restrict__ agent, const u16* __restrict__ qkv,
    const float* __restrict__ pb, float* __restrict__ s1)
{
  int nt = blockIdx.x, bh = blockIdx.y;
  int b = bh >> 3, h = bh & 7;
  int lane = threadIdx.x & 31;
  int wv   = threadIdx.x >> 5;
  const u16* aBase = agent + (size_t)(bh * AGP + wv * 16) * DH;
  const u16* kBase = qkv + (size_t)(b * NTOK + nt * 16) * NQKV + IDIM + h * DH;
  v16bf a0 = loadAfrag(aBase, DH, lane, 0);
  v16bf a1 = loadAfrag(aBase, DH, lane, 32);
  v16bf k0 = loadBfragT(kBase, NQKV, lane, 0);
  v16bf k1 = loadBfragT(kBase, NQKV, lane, 32);
  v8f acc = {};
  acc = wmma_bf16(a0, k0, acc);
  acc = wmma_bf16(a1, k1, acc);
  int n    = nt * 16 + (lane & 15);
  int aloc = wv * 16 + ((lane >> 4) << 3);
  #pragma unroll
  for (int r = 0; r < 8; ++r) {
    int a = aloc + r;
    s1[(size_t)(bh * AGP + a) * NTOK + n] =
        acc[r] + pb[(size_t)(h * AGP + a) * NTOK + n];
  }
}

// ----------------------- softmax over n, emit bf16 -------------------------

__global__ __launch_bounds__(256) void softmax_rows(
    const float* __restrict__ s1, u16* __restrict__ attn)
{
  __shared__ float red[256];
  int row = blockIdx.x;                    // bh*64 + a
  const float* p = s1 + (size_t)row * NTOK;
  int t = threadIdx.x;
  float mx = -1e30f;
  for (int i = t; i < NTOK; i += 256) mx = fmaxf(mx, p[i]);
  red[t] = mx; __syncthreads();
  for (int s = 128; s > 0; s >>= 1) {
    if (t < s) red[t] = fmaxf(red[t], red[t + s]);
    __syncthreads();
  }
  mx = red[0]; __syncthreads();
  float sm = 0.f;
  for (int i = t; i < NTOK; i += 256) sm += __expf(p[i] - mx);
  red[t] = sm; __syncthreads();
  for (int s = 128; s > 0; s >>= 1) {
    if (t < s) red[t] += red[t + s];
    __syncthreads();
  }
  float inv = 1.f / red[0];
  u16* o = attn + (size_t)row * NTOK;
  for (int i = t; i < NTOK; i += 256) o[i] = f2bf(__expf(p[i] - mx) * inv);
}

// ------------------- v transpose: vt[b][h][d][n] (bf16) --------------------

__global__ void transpose_v(const u16* __restrict__ qkv, u16* __restrict__ vt)
{
  int d  = blockIdx.x & 63;
  int bh = blockIdx.x >> 6;
  int b = bh >> 3, h = bh & 7;
  const u16* src = qkv + (size_t)b * NTOK * NQKV + 2 * IDIM + h * DH + d;
  u16* dst = vt + (size_t)(bh * DH + d) * NTOK;
  for (int n = threadIdx.x; n < NTOK; n += blockDim.x)
    dst[n] = src[(size_t)n * NQKV];
}

// ------------- agent_v = attn @ v ; stored transposed avt[d][a] ------------
// grid (b*h); 8 waves cover 16 (a-tile, d-tile) pairs; K = 3136.

__global__ __launch_bounds__(256) void agent_v_gemm(
    const u16* __restrict__ attn, const u16* __restrict__ vt,
    u16* __restrict__ avt)
{
  int bh   = blockIdx.x;
  int lane = threadIdx.x & 31;
  int wv   = threadIdx.x >> 5;
  for (int t = wv; t < 16; t += 8) {
    int mt = t >> 2, dt = t & 3;
    const u16* aBase = attn + (size_t)(bh * AGP + mt * 16) * NTOK;
    const u16* bBase = vt + (size_t)(bh * DH + dt * 16) * NTOK;
    v8f acc = {};
    #pragma unroll 2
    for (int k0 = 0; k0 < NTOK; k0 += 32) {
      v16bf av = loadAfrag(aBase, NTOK, lane, k0);
      v16bf bv = loadBfragT(bBase, NTOK, lane, k0);
      acc = wmma_bf16(av, bv, acc);
    }
    int d  = dt * 16 + (lane & 15);
    int al = mt * 16 + ((lane >> 4) << 3);
    #pragma unroll
    for (int r = 0; r < 8; ++r)
      avt[(size_t)(bh * DH + d) * AGP + al + r] = f2bf(acc[r]);
  }
}

// ------ fused: s2 = q @ (scale*agent)^T + nb ; softmax(a) ; out = qa@av -----
// grid (196 ntiles, b*h); 4 waves: wave = a-tile in phase1, d-tile in phase3.

__global__ __launch_bounds__(128) void attn2_out(
    const u16* __restrict__ qkv, const u16* __restrict__ agent,
    const float* __restrict__ nb, const u16* __restrict__ avt,
    float* __restrict__ outp)
{
  __shared__ float s2s[16][64];
  __shared__ __align__(16) u16 qat[16][64];
  __shared__ float red[16][8];
  __shared__ float rmx[16];
  __shared__ float rsum[16];

  int nt = blockIdx.x, bh = blockIdx.y;
  int b = bh >> 3, h = bh & 7;
  int lane = threadIdx.x & 31;
  int wv   = threadIdx.x >> 5;

  // phase 1: 16(n) x 16(a) logits per wave
  {
    const u16* aBase = qkv + (size_t)(b * NTOK + nt * 16) * NQKV + h * DH;
    const u16* bBase = agent + (size_t)(bh * AGP + wv * 16) * DH;
    v16bf a0 = loadAfrag(aBase, NQKV, lane, 0);
    v16bf a1 = loadAfrag(aBase, NQKV, lane, 32);
    v16bf b0 = loadBfragT(bBase, DH, lane, 0);
    v16bf b1 = loadBfragT(bBase, DH, lane, 32);
    v8f acc = {};
    acc = wmma_bf16(a0, b0, acc);
    acc = wmma_bf16(a1, b1, acc);
    int aC = wv * 16 + (lane & 15);
    int nl = (lane >> 4) << 3;
    #pragma unroll
    for (int r = 0; r < 8; ++r) {
      int n = nt * 16 + nl + r;
      s2s[nl + r][aC] = acc[r] + nb[(size_t)(h * NTOK + n) * AGP + aC];
    }
  }
  __syncthreads();
  // phase 2: softmax over a=64 per row (8 threads/row)
  {
    int row = threadIdx.x >> 3;
    int seg = threadIdx.x & 7;
    float m = -1e30f;
    #pragma unroll
    for (int i = 0; i < 8; ++i) m = fmaxf(m, s2s[row][seg * 8 + i]);
    red[row][seg] = m;
    __syncthreads();
    if (seg == 0) {
      float mm = red[row][0];
      for (int i = 1; i < 8; ++i) mm = fmaxf(mm, red[row][i]);
      rmx[row] = mm;
    }
    __syncthreads();
    float mm = rmx[row];
    float s = 0.f;
    #pragma unroll
    for (int i = 0; i < 8; ++i) s += __expf(s2s[row][seg * 8 + i] - mm);
    red[row][seg] = s;
    __syncthreads();
    if (seg == 0) {
      float ss = 0.f;
      for (int i = 0; i < 8; ++i) ss += red[row][i];
      rsum[row] = ss;
    }
    __syncthreads();
    float inv = 1.f / rsum[row];
    #pragma unroll
    for (int i = 0; i < 8; ++i)
      qat[row][seg * 8 + i] = f2bf(__expf(s2s[row][seg * 8 + i] - mm) * inv);
  }
  __syncthreads();
  // phase 3: out[nx16d] = qat(16x64) @ agent_v(64x64), B from avt[d][a]
  {
    const u16* bBase = avt + (size_t)(bh * DH + wv * 16) * AGP;
    v16bf a0 = loadAfrag(&qat[0][0], 64, lane, 0);
    v16bf a1 = loadAfrag(&qat[0][0], 64, lane, 32);
    v16bf b0 = loadBfragT(bBase, AGP, lane, 0);
    v16bf b1 = loadBfragT(bBase, AGP, lane, 32);
    v8f acc = {};
    acc = wmma_bf16(a0, b0, acc);
    acc = wmma_bf16(a1, b1, acc);
    int d  = wv * 16 + (lane & 15);
    int nl = (lane >> 4) << 3;
    #pragma unroll
    for (int r = 0; r < 8; ++r) {
      int n = nt * 16 + nl + r;
      outp[(size_t)(b * NTOK + n) * IDIM + h * DH + d] = acc[r];
    }
  }
}

// -------- depthwise 3x3 conv on v, add bias + attention out -> bf16 --------

__global__ void dwc_add(const u16* __restrict__ qkv, const float* __restrict__ dww,
                        const float* __restrict__ dwb, const float* __restrict__ outp,
                        u16* __restrict__ obf)
{
  size_t gid = (size_t)blockIdx.x * blockDim.x + threadIdx.x;
  if (gid >= (size_t)MROWS * IDIM) return;
  int c = gid % IDIM;
  size_t rem = gid / IDIM;
  int x = rem % WW; rem /= WW;
  int y = rem % HH;
  int b = rem / HH;
  float acc = 0.f;
  #pragma unroll
  for (int ky = 0; ky < 3; ++ky) {
    int yy = y + ky - 1;
    if (yy < 0 || yy >= HH) continue;
    #pragma unroll
    for (int kx = 0; kx < 3; ++kx) {
      int xx = x + kx - 1;
      if (xx < 0 || xx >= WW) continue;
      acc += dww[c * 9 + ky * 3 + kx] *
             bf2f(qkv[(size_t)(b * NTOK + yy * WW + xx) * NQKV + 2 * IDIM + c]);
    }
  }
  size_t oi = (size_t)(b * NTOK + y * WW + x) * IDIM + c;
  obf[oi] = f2bf(acc + dwb[c] + outp[oi]);
}

// ---------------------------------------------------------------------------

extern "C" void kernel_launch(void* const* d_in, const int* in_sizes, int n_in,
                              void* d_out, int out_size, void* d_ws, size_t ws_size,
                              hipStream_t stream)
{
  (void)in_sizes; (void)n_in; (void)out_size; (void)ws_size;
  const float* x    = (const float*)d_in[0];
  const float* qw   = (const float*)d_in[3];
  const float* qb   = (const float*)d_in[4];
  const float* kvw  = (const float*)d_in[5];
  const float* kvb  = (const float*)d_in[6];
  const float* pw   = (const float*)d_in[7];
  const float* pjb  = (const float*)d_in[8];
  const float* dww  = (const float*)d_in[9];
  const float* dwb  = (const float*)d_in[10];
  const float* anb  = (const float*)d_in[11];
  const float* nab  = (const float*)d_in[12];
  const float* ahb  = (const float*)d_in[13];
  const float* awb  = (const float*)d_in[14];
  const float* hab  = (const float*)d_in[15];
  const float* wab  = (const float*)d_in[16];

  char* ws = (char*)d_ws;
  size_t off = 0;
  auto alloc = [&](size_t bytes) -> void* {
    void* p = ws + off;
    off = (off + bytes + 255) & ~(size_t)255;
    return p;
  };

  u16*   XB    = (u16*)  alloc((size_t)MROWS * DIM * 2);            // 51.4 MB
  u16*   WQKV  = (u16*)  alloc((size_t)DIM * NQKV * 2);
  u16*   WPROJ = (u16*)  alloc((size_t)DIM * DIM * 2);
  float* BQKV  = (float*)alloc((size_t)NQKV * 4);
  u16*   QKV   = (u16*)  alloc((size_t)MROWS * NQKV * 2);           // 154 MB
  u16*   AGENT = (u16*)  alloc((size_t)BATCH * HEADS * AGP * DH * 2);
  float* PB    = (float*)alloc((size_t)HEADS * AGP * NTOK * 4);
  float* NB    = (float*)alloc((size_t)HEADS * NTOK * AGP * 4);
  float* S1    = (float*)alloc((size_t)BATCH * HEADS * AGP * NTOK * 4); // 103 MB
  u16*   VT    = (u16*)  alloc((size_t)BATCH * HEADS * DH * NTOK * 2); // 51.4 MB
  u16*   AVT   = (u16*)  alloc((size_t)BATCH * HEADS * DH * AGP * 2);
  // lifetime-based aliases (sizes match exactly):
  u16*   ATTN  = XB;   // xb dead after qkv GEMM
  float* OUTP  = S1;   // s1 dead after softmax
  u16*   OBF   = VT;   // vt dead after agent_v GEMM

  dim3 b256(256), b128(128);

  size_t nx = (size_t)MROWS * DIM;
  pack_x<<<(unsigned)((nx + 255) / 256), b256, 0, stream>>>(x, XB, nx);
  pack_w<<<(DIM * NQKV + 255) / 256, b256, 0, stream>>>(qw, qb, kvw, kvb, pw,
                                                        WQKV, WPROJ, BQKV);
  gemm64<NQKV, DIM, false><<<dim3(MROWS / 64, NQKV / 64), b128, 0, stream>>>(
      XB, WQKV, BQKV, QKV);

  agent_pool<<<BATCH * AGP, b256, 0, stream>>>(QKV, AGENT);
  bias_pb<<<HEADS * AGP, b256, 0, stream>>>(anb, ahb, awb, PB);
  bias_nb<<<HEADS * NTOK, dim3(64), 0, stream>>>(nab, hab, wab, NB);

  attn_s1<<<dim3(NTOK / 16, BATCH * HEADS), b128, 0, stream>>>(AGENT, QKV, PB, S1);
  transpose_v<<<BATCH * HEADS * DH, b256, 0, stream>>>(QKV, VT);
  softmax_rows<<<BATCH * HEADS * AGP, b256, 0, stream>>>(S1, ATTN);
  agent_v_gemm<<<BATCH * HEADS, b256, 0, stream>>>(ATTN, VT, AVT);
  attn2_out<<<dim3(NTOK / 16, BATCH * HEADS), b128, 0, stream>>>(
      QKV, AGENT, NB, AVT, OUTP);

  size_t tot = (size_t)MROWS * IDIM;
  dwc_add<<<(unsigned)((tot + 255) / 256), b256, 0, stream>>>(QKV, dww, dwb,
                                                              OUTP, OBF);
  gemm64<DIM, DIM, true><<<dim3(MROWS / 64, DIM / 64), b128, 0, stream>>>(
      OBF, WPROJ, pjb, (float*)d_out);
}
